// AttentionBlock_82068235092292
// MI455X (gfx1250) — matrix operand order repaired and checked by reading, not compiled, not run on hardware
//
#include <hip/hip_runtime.h>
#include <math.h>

typedef __attribute__((ext_vector_type(16))) _Float16 v16h;
typedef __attribute__((ext_vector_type(8)))  float    v8f;
typedef __attribute__((ext_vector_type(4)))  int      v4i;

#define DIMC 512
#define HEADS 8
#define HDIM 64
#define BATCH 4
#define HW 1024
#define NTOK 4096
#define TXTLEN 77
#define LATD 512
#define NEXP 8
#define HIDD 2048

#if __has_builtin(__builtin_amdgcn_global_load_async_to_lds_b128)
#define HAVE_ASYNC_LDS 1
#else
#define HAVE_ASYNC_LDS 0
#endif

#define AS1 __attribute__((address_space(1)))
#define AS3 __attribute__((address_space(3)))

static __device__ __forceinline__ void async_wait0() {
#if HAVE_ASYNC_LDS
#if __has_builtin(__builtin_amdgcn_s_wait_asynccnt)
  __builtin_amdgcn_s_wait_asynccnt(0);
#else
  asm volatile("s_wait_asynccnt 0" ::: "memory");
#endif
#endif
}

// copy 16B global(f16) -> LDS via async DMA when available
static __device__ __forceinline__ void cp_async16(_Float16* ldst, const _Float16* gsrc) {
#if HAVE_ASYNC_LDS
  __builtin_amdgcn_global_load_async_to_lds_b128(
      (AS1 v4i*)gsrc, (AS3 v4i*)ldst, 0, 0);
#else
  *(uint4*)ldst = *(const uint4*)gsrc;
#endif
}

// convert 16 f32 (4 x float4) -> 16 f16 and store 32B to LDS
static __device__ __forceinline__ void cvt16_store(_Float16* dst, const float4* f) {
  union { _Float16 h[16]; uint4 u[2]; } t;
#pragma unroll
  for (int i = 0; i < 4; ++i) {
    t.h[4 * i + 0] = (_Float16)f[i].x;
    t.h[4 * i + 1] = (_Float16)f[i].y;
    t.h[4 * i + 2] = (_Float16)f[i].z;
    t.h[4 * i + 3] = (_Float16)f[i].w;
  }
  ((uint4*)dst)[0] = t.u[0];
  ((uint4*)dst)[1] = t.u[1];
}

// ---------------------------------------------------------------------------
// WMMA fragment loaders (CDNA5 wave32 layouts, CDNA5 ISA §7.12.2)
// ---------------------------------------------------------------------------
static __device__ __forceinline__ v16h frag_A_lds(const _Float16* base, int m0, int ldh) {
  int L = threadIdx.x & 31;
  const unsigned* row = (const unsigned*)(base + (unsigned)(m0 + (L & 15)) * ldh);
  int off = (L >> 4) * 4;
  union { v16h h; unsigned u[8]; } r;
#pragma unroll
  for (int v = 0; v < 4; ++v) { r.u[v] = row[off + v]; r.u[v + 4] = row[off + 8 + v]; }
  return r.h;
}

static __device__ __forceinline__ v16h frag_B_lds(const _Float16* base, int n0, int ldh, int koff_dw) {
  int L = threadIdx.x & 31;
  const unsigned* row = (const unsigned*)(base + (unsigned)(n0 + (L & 15)) * ldh);
  int off = koff_dw + (L >> 4) * 4;
  union { v16h h; unsigned u[8]; } r;
#pragma unroll
  for (int v = 0; v < 4; ++v) { r.u[v] = row[off + v]; r.u[v + 4] = row[off + 8 + v]; }
  return r.h;
}

static __device__ __forceinline__ v8f wmma_f16(v16h a, v16h b, v8f c) {
  return __builtin_amdgcn_wmma_f32_16x16x32_f16(false, a, false, b, (short)0, c, false, false);
}

// ---------------------------------------------------------------------------
// Generic tiled WMMA GEMM:  C(MxN) = act(A(MxK,f32) * B(f16,[N,K]) + bias) [+C]
// Block = 128 threads (4 waves); tile 64x64; K step 32.
// B tiles staged with GLOBAL_LOAD_ASYNC_TO_LDS_B128 (byte-identical layout).
//  ACOL : A element [m][k] at A[k*lda+m]      CCOL : store C[m][n] at C[n*ldc+m]
//  AMAP : gather A rows via rowmap            CMAP : scatter C rows via rowmap
//  Mdev : optional device row count (MoE expert counts)
// ---------------------------------------------------------------------------
template<bool ACOL, bool GELU_ACT, bool ADD, bool CCOL, bool AMAP, bool CMAP>
__global__ void __launch_bounds__(128)
gemm_k(const float* __restrict__ A, int lda, long a_bs,
       const _Float16* __restrict__ Bh, int ldb, long b_bs,
       const float* __restrict__ bias,
       float* __restrict__ C, int ldc, long c_bs,
       int M, int K,
       const int* __restrict__ Mdev, const int* __restrict__ rowmap)
{
  __shared__ _Float16 As[64 * 32];
  __shared__ _Float16 Bs[64 * 32];
  int Mlim = Mdev ? Mdev[0] : M;
  int m0 = blockIdx.y * 64;
  if (m0 >= Mlim) return;
  int n0 = blockIdx.x * 64;
  A += (long)blockIdx.z * a_bs;
  Bh += (long)blockIdx.z * b_bs;
  C += (long)blockIdx.z * c_bs;
  int tid = threadIdx.x;
  int wave = tid >> 5;

  v8f z8 = {0.f,0.f,0.f,0.f,0.f,0.f,0.f,0.f};
  v8f acc[4] = {z8, z8, z8, z8};

  int srow = tid >> 1;            // 0..63
  int skh  = (tid & 1) * 16;      // 0 / 16 (halves)
  int ar = m0 + srow;
  bool av = ar < Mlim;
  int arow = av ? (AMAP ? rowmap[ar] : ar) : 0;
  const _Float16* bsrc = Bh + (long)(n0 + srow) * ldb + skh;
  _Float16* bdst = Bs + srow * 32 + skh;
  _Float16* adst = As + srow * 32 + skh;

  for (int k0 = 0; k0 < K; k0 += 32) {
    // ---- B tile: f16 [N,K] rows are contiguous -> async DMA straight to LDS
    cp_async16(bdst, bsrc + k0);
    cp_async16(bdst + 8, bsrc + k0 + 8);
    // ---- A tile: f32 -> f16 conversion staging (vectorized when row-major)
    if (ACOL) {
      union { _Float16 h[16]; uint4 u[2]; } t;
#pragma unroll
      for (int j = 0; j < 16; ++j) {
        int kk = k0 + skh + j;
        t.h[j] = (_Float16)(av ? A[(long)kk * lda + arow] : 0.0f);
      }
      ((uint4*)adst)[0] = t.u[0];
      ((uint4*)adst)[1] = t.u[1];
    } else {
      float4 f[4] = {{0,0,0,0},{0,0,0,0},{0,0,0,0},{0,0,0,0}};
      if (av) {
        const float4* ap = (const float4*)(A + (long)arow * lda + k0 + skh);
#pragma unroll
        for (int i = 0; i < 4; ++i) f[i] = ap[i];
        if (k0 + 32 < K) __builtin_prefetch(A + (long)arow * lda + k0 + 32 + skh, 0, 1);
      }
      cvt16_store(adst, f);
    }
    async_wait0();
    __syncthreads();
    v16h af = frag_A_lds(As, wave * 16, 32);
#pragma unroll
    for (int s = 0; s < 4; ++s) {
      v16h bf = frag_B_lds(Bs, s * 16, 32, 0);
      acc[s] = wmma_f16(af, bf, acc[s]);
    }
    __syncthreads();
  }

  int L = tid & 31;
  int cn = L & 15;
  int rbase = m0 + wave * 16 + ((L >> 4) * 8);
#pragma unroll
  for (int s = 0; s < 4; ++s) {
    int col = n0 + s * 16 + cn;
    float bb = bias ? bias[col] : 0.0f;
#pragma unroll
    for (int r = 0; r < 8; ++r) {
      int rowi = rbase + r;
      if (rowi < Mlim) {
        float val = acc[s][r] + bb;
        if (GELU_ACT) val = 0.5f * val * (1.0f + erff(val * 0.70710678118f));
        int crow = CMAP ? rowmap[rowi] : rowi;
        long idx = CCOL ? ((long)col * ldc + crow) : ((long)crow * ldc + col);
        if (ADD) C[idx] += val; else C[idx] = val;
      }
    }
  }
}

// ---------------------------------------------------------------------------
// Flash attention: 4 waves x 16 q-rows = 64 queries/block, hd=64.
// ---------------------------------------------------------------------------
__global__ void __launch_bounds__(128)
attn_k(const float* __restrict__ Q, int q_rs, long q_bs, int q_base,
       const float* __restrict__ Km, int k_rs, long k_bs, int k_base,
       const float* __restrict__ Vm, int v_rs, long v_bs, int v_base,
       float* __restrict__ O, int kv_len, float scale)
{
  __shared__ _Float16 Ks[32 * 64];      // [key][dim]
  __shared__ _Float16 Vs[64 * 32];      // [dim][key]
  __shared__ _Float16 Ps[4][16 * 32];   // per-wave P tile
  int bh = blockIdx.y;
  int b = bh >> 3, h = bh & 7;
  int q0 = blockIdx.x * 64;
  int tid = threadIdx.x;
  int wave = tid >> 5;
  int L = tid & 31;
  const float* qp = Q + (long)b * q_bs + q_base + h * HDIM;
  const float* kp = Km + (long)b * k_bs + k_base + h * HDIM;
  const float* vp = Vm + (long)b * v_bs + v_base + h * HDIM;

  // Q fragments: per lane, halves 0..7 = qrow[c*32+kb+0..7], 8..15 = qrow[c*32+kb+16..23]
  int qm = q0 + wave * 16 + (L & 15);
  int kb = (L >> 4) * 8;
  v16h qf[2];
#pragma unroll
  for (int c = 0; c < 2; ++c) {
    const float4* qrow = (const float4*)(qp + (long)qm * q_rs + c * 32 + kb);
    float4 f[4];
    f[0] = qrow[0]; f[1] = qrow[1]; f[2] = qrow[4]; f[3] = qrow[5];
    union { v16h v; _Float16 h[16]; uint4 u[2]; } r;
    union { _Float16 h[16]; uint4 u[2]; } t;
#pragma unroll
    for (int i = 0; i < 4; ++i) {
      t.h[4 * i + 0] = (_Float16)f[i].x;
      t.h[4 * i + 1] = (_Float16)f[i].y;
      t.h[4 * i + 2] = (_Float16)f[i].z;
      t.h[4 * i + 3] = (_Float16)f[i].w;
    }
    r.u[0] = t.u[0]; r.u[1] = t.u[1];
    qf[c] = r.v;
  }

  v8f z8 = {0.f,0.f,0.f,0.f,0.f,0.f,0.f,0.f};
  v8f oacc[4] = {z8, z8, z8, z8};
  float mrow[8], lrow[8];
#pragma unroll
  for (int r = 0; r < 8; ++r) { mrow[r] = -__builtin_inff(); lrow[r] = 0.0f; }

  int nchunks = (kv_len + 31) / 32;
  for (int kc = 0; kc < nchunks; ++kc) {
    int key = tid >> 2;
    int dh  = (tid & 3) * 16;
    int kg  = kc * 32 + key;
    bool kvld = kg < kv_len;
    {
      float4 fk[4] = {{0,0,0,0},{0,0,0,0},{0,0,0,0},{0,0,0,0}};
      if (kvld) {
        const float4* kr = (const float4*)(kp + (long)kg * k_rs + dh);
#pragma unroll
        for (int i = 0; i < 4; ++i) fk[i] = kr[i];
      }
      cvt16_store(Ks + key * 64 + dh, fk);
      float4 fv[4] = {{0,0,0,0},{0,0,0,0},{0,0,0,0},{0,0,0,0}};
      if (kvld) {
        const float4* vr = (const float4*)(vp + (long)kg * v_rs + dh);
#pragma unroll
        for (int i = 0; i < 4; ++i) fv[i] = vr[i];
      }
#pragma unroll
      for (int i = 0; i < 4; ++i) {
        Vs[(dh + 4 * i + 0) * 32 + key] = (_Float16)fv[i].x;
        Vs[(dh + 4 * i + 1) * 32 + key] = (_Float16)fv[i].y;
        Vs[(dh + 4 * i + 2) * 32 + key] = (_Float16)fv[i].z;
        Vs[(dh + 4 * i + 3) * 32 + key] = (_Float16)fv[i].w;
      }
    }
    __syncthreads();

    // S = Q K^T over hd=64: 2 key-subtiles x 2 contraction chunks
    v8f sacc[2] = {z8, z8};
#pragma unroll
    for (int c = 0; c < 2; ++c)
#pragma unroll
      for (int s = 0; s < 2; ++s) {
        v16h bf = frag_B_lds(Ks, s * 16, 64, c * 16);
        sacc[s] = wmma_f16(qf[c], bf, sacc[s]);
      }

    float p[2][8];
#pragma unroll
    for (int s = 0; s < 2; ++s) {
      int keyg = kc * 32 + s * 16 + (L & 15);
      bool valid = keyg < kv_len;
#pragma unroll
      for (int r = 0; r < 8; ++r)
        p[s][r] = valid ? sacc[s][r] * scale : -__builtin_inff();
    }

    // online softmax (rows per-VGPR; cols across the 16-lane half)
#pragma unroll
    for (int r = 0; r < 8; ++r) {
      float x = fmaxf(p[0][r], p[1][r]);
#pragma unroll
      for (int d = 1; d < 16; d <<= 1) x = fmaxf(x, __shfl_xor(x, d));
      float mnew = fmaxf(mrow[r], x);
      float corr = __expf(mrow[r] - mnew);
      p[0][r] = __expf(p[0][r] - mnew);
      p[1][r] = __expf(p[1][r] - mnew);
      float ls = p[0][r] + p[1][r];
#pragma unroll
      for (int d = 1; d < 16; d <<= 1) ls += __shfl_xor(ls, d);
      lrow[r] = lrow[r] * corr + ls;
      mrow[r] = mnew;
#pragma unroll
      for (int ns = 0; ns < 4; ++ns) oacc[ns][r] *= corr;
    }

    // spill P (C-layout) to per-wave LDS, reload as A-fragment
    _Float16* pw = &Ps[wave][0];
#pragma unroll
    for (int s = 0; s < 2; ++s)
#pragma unroll
      for (int r = 0; r < 8; ++r)
        pw[(r + (L >> 4) * 8) * 32 + s * 16 + (L & 15)] = (_Float16)p[s][r];
    asm volatile("s_wait_dscnt 0" ::: "memory");
    v16h pf = frag_A_lds(pw, 0, 32);
#pragma unroll
    for (int ns = 0; ns < 4; ++ns) {
      v16h vf = frag_B_lds(Vs, ns * 16, 32, 0);
      oacc[ns] = wmma_f16(pf, vf, oacc[ns]);
    }
    __syncthreads();
  }

  int token = b * HW + q0 + wave * 16 + ((L >> 4) * 8);
#pragma unroll
  for (int ns = 0; ns < 4; ++ns) {
    int col = h * HDIM + ns * 16 + (L & 15);
#pragma unroll
    for (int r = 0; r < 8; ++r)
      O[(long)(token + r) * DIMC + col] = oacc[ns][r] / lrow[r];
  }
}

// ---------------------------------------------------------------------------
// LayerNorm over 512, one block (128 threads) per token.
// ---------------------------------------------------------------------------
__global__ void __launch_bounds__(128)
ln_k(const float* __restrict__ X, const float* __restrict__ g,
     const float* __restrict__ bta, float* __restrict__ Y)
{
  __shared__ float red[4];
  int t = blockIdx.x, tid = threadIdx.x;
  const float* xr = X + (long)t * DIMC;
  float v[4]; float s = 0.f;
#pragma unroll
  for (int j = 0; j < 4; ++j) { v[j] = xr[tid + j * 128]; s += v[j]; }
#pragma unroll
  for (int d = 1; d < 32; d <<= 1) s += __shfl_xor(s, d);
  if ((tid & 31) == 0) red[tid >> 5] = s;
  __syncthreads();
  float mu = (red[0] + red[1] + red[2] + red[3]) * (1.0f / DIMC);
  __syncthreads();
  float q = 0.f;
#pragma unroll
  for (int j = 0; j < 4; ++j) { float d2 = v[j] - mu; q += d2 * d2; }
#pragma unroll
  for (int d = 1; d < 32; d <<= 1) q += __shfl_xor(q, d);
  if ((tid & 31) == 0) red[tid >> 5] = q;
  __syncthreads();
  float var = (red[0] + red[1] + red[2] + red[3]) * (1.0f / DIMC);
  float rs = rsqrtf(var + 1e-5f);
#pragma unroll
  for (int j = 0; j < 4; ++j) {
    int c = tid + j * 128;
    Y[(long)t * DIMC + c] = (v[j] - mu) * rs * g[c] + bta[c];
  }
}

// ---------------------------------------------------------------------------
// ModulatedConv helpers: style = w @ mod_w.T + mod_b ; demodulated f16 weights.
// ---------------------------------------------------------------------------
__global__ void __launch_bounds__(512)
style_k(const float* __restrict__ w, const float* __restrict__ mw,
        const float* __restrict__ mb, float* __restrict__ style)
{
  int b = blockIdx.x, c = threadIdx.x;
  const float* wr = w + b * LATD;
  const float* mr = mw + (long)c * LATD;
  float s = 0.f;
  for (int l = 0; l < LATD; ++l) s += wr[l] * mr[l];
  style[b * DIMC + c] = s + mb[c];
}

__global__ void __launch_bounds__(256)
modw_k(const float* __restrict__ weight, const float* __restrict__ style,
       _Float16* __restrict__ wt)
{
  __shared__ float red[8];
  int o = blockIdx.x, b = blockIdx.y, tid = threadIdx.x;
  const float* wrow = weight + (long)o * DIMC;
  const float* srow = style + b * DIMC;
  float t0 = wrow[tid] * srow[tid];
  float t1 = wrow[tid + 256] * srow[tid + 256];
  float q = t0 * t0 + t1 * t1;
#pragma unroll
  for (int d = 1; d < 32; d <<= 1) q += __shfl_xor(q, d);
  if ((tid & 31) == 0) red[tid >> 5] = q;
  __syncthreads();
  float sum = 0.f;
#pragma unroll
  for (int i = 0; i < 8; ++i) sum += red[i];
  float sc = rsqrtf(sum + 1e-8f);
  _Float16* out = wt + ((long)b * DIMC + o) * DIMC;
  out[tid] = (_Float16)(t0 * sc);
  out[tid + 256] = (_Float16)(t1 * sc);
}

// ---------------------------------------------------------------------------
// Weight pre-conversion: f32 -> f16 (straight, vectorized) and transposed.
// ---------------------------------------------------------------------------
__global__ void __launch_bounds__(256)
cvt16_k(const float* __restrict__ s, _Float16* __restrict__ d, long n)
{
  long i = ((long)blockIdx.x * 256 + threadIdx.x) * 4;
  if (i + 3 < n) {
    float4 f = *(const float4*)(s + i);
    union { _Float16 h[4]; uint2 u; } t;
    t.h[0] = (_Float16)f.x; t.h[1] = (_Float16)f.y;
    t.h[2] = (_Float16)f.z; t.h[3] = (_Float16)f.w;
    *(uint2*)(d + i) = t.u;
  }
}

__global__ void __launch_bounds__(256)
cvtT_k(const float* __restrict__ s, _Float16* __restrict__ d, int K, int N)
{
  long i = (long)blockIdx.x * 256 + threadIdx.x;
  if (i < (long)K * N) {
    int k = (int)(i / N), n = (int)(i % N);
    d[(long)n * K + k] = (_Float16)s[i];
  }
}

// ---------------------------------------------------------------------------
// Router: logits argmax (positive clipped temperature cannot change argmax).
// ---------------------------------------------------------------------------
__global__ void __launch_bounds__(128)
router_k(const float* __restrict__ tok, const float* __restrict__ wlat,
         const float* __restrict__ fmu, const float* __restrict__ tmu,
         const float* __restrict__ cmu,
         float* __restrict__ onehot, int* __restrict__ eid)
{
  __shared__ float comb[256];
  __shared__ float lg[8];
  int t = blockIdx.x, j = threadIdx.x;
  int b = t >> 10;
  const float* tr = tok + (long)t * DIMC;
  const float* wr = wlat + b * LATD;
  float a0 = 0.f, a1 = 0.f;
  for (int k = 0; k < DIMC; ++k) {
    a0 += tr[k] * fmu[k * 128 + j];
    a1 += wr[k] * tmu[k * 128 + j];
  }
  comb[j] = a0;
  comb[128 + j] = a1;
  __syncthreads();
  if (j < 8) {
    float s = 0.f;
    for (int i = 0; i < 256; ++i) s += comb[i] * cmu[i * NEXP + j];
    lg[j] = s;
  }
  __syncthreads();
  if (j == 0) {
    int am = 0; float mx = lg[0];
    for (int e = 1; e < 8; ++e) if (lg[e] > mx) { mx = lg[e]; am = e; }
    eid[t] = am;
    float* oh = onehot + (long)t * NEXP;
    for (int e = 0; e < NEXP; ++e) oh[e] = (e == am) ? 1.0f : 0.0f;
  }
}

__global__ void zero8_k(int* cnt) { if (threadIdx.x < 8) cnt[threadIdx.x] = 0; }

__global__ void scatter_k(const int* __restrict__ eid, int* cnt, int* lists) {
  int t = blockIdx.x * blockDim.x + threadIdx.x;
  if (t < NTOK) {
    int e = eid[t];
    int p = atomicAdd(&cnt[e], 1);
    lists[e * NTOK + p] = t;
  }
}

// ---------------------------------------------------------------------------
extern "C" void kernel_launch(void* const* d_in, const int* in_sizes, int n_in,
                              void* d_out, int out_size, void* d_ws, size_t ws_size,
                              hipStream_t stream) {
  const float* x        = (const float*)d_in[0];
  const float* w        = (const float*)d_in[1];
  const float* text_seq = (const float*)d_in[2];
  const float* pin_w    = (const float*)d_in[3];
  const float* pin_mw   = (const float*)d_in[4];
  const float* pin_mb   = (const float*)d_in[5];
  const float* pout_w   = (const float*)d_in[6];
  const float* pout_mw  = (const float*)d_in[7];
  const float* pout_mb  = (const float*)d_in[8];
  const float* ln1_g = (const float*)d_in[9],  *ln1_b = (const float*)d_in[10];
  const float* ln2_g = (const float*)d_in[11], *ln2_b = (const float*)d_in[12];
  const float* ln3_g = (const float*)d_in[13], *ln3_b = (const float*)d_in[14];
  const float* sa_in_w  = (const float*)d_in[15], *sa_in_b  = (const float*)d_in[16];
  const float* sa_out_w = (const float*)d_in[17], *sa_out_b = (const float*)d_in[18];
  const float* ca_in_w  = (const float*)d_in[19], *ca_in_b  = (const float*)d_in[20];
  const float* ca_out_w = (const float*)d_in[21], *ca_out_b = (const float*)d_in[22];
  const float* r_feat = (const float*)d_in[23];
  const float* r_text = (const float*)d_in[24];
  const float* r_comb = (const float*)d_in[25];
  const float* e_w1 = (const float*)d_in[27], *e_b1 = (const float*)d_in[28];
  const float* e_w2 = (const float*)d_in[29], *e_b2 = (const float*)d_in[30];

  float* ws = (float*)d_ws;
  size_t off = 0;
  auto alloc = [&](size_t n) { float* p = ws + off; off += n; return p; };
  float* style_in  = alloc(BATCH * DIMC);
  float* style_out = alloc(BATCH * DIMC);
  float* x_flat = alloc((size_t)NTOK * DIMC);
  float* xn     = alloc((size_t)NTOK * DIMC);
  float* qkv    = alloc((size_t)NTOK * 3 * DIMC);
  float* attn_o = alloc((size_t)NTOK * DIMC);
  float* hid    = alloc((size_t)NTOK * HIDD);
  // f16 weight mirrors (allocated in half units; region starts 16B-aligned)
  _Float16* hbase = (_Float16*)(ws + off);
  size_t hoff = 0;
  auto halloc = [&](size_t n) { _Float16* p = hbase + hoff; hoff += n; return p; };
  _Float16* wt16_in  = halloc((size_t)BATCH * DIMC * DIMC);
  _Float16* wt16_out = halloc((size_t)BATCH * DIMC * DIMC);
  _Float16* saw16  = halloc((size_t)3 * DIMC * DIMC);
  _Float16* saow16 = halloc((size_t)DIMC * DIMC);
  _Float16* caw16  = halloc((size_t)3 * DIMC * DIMC);
  _Float16* caow16 = halloc((size_t)DIMC * DIMC);
  _Float16* ew1t = halloc((size_t)NEXP * HIDD * DIMC);  // [e][N=2048][K=512]
  _Float16* ew2t = halloc((size_t)NEXP * DIMC * HIDD);  // [e][N=512][K=2048]
  off += (hoff + 1) / 2;
  int* eid   = (int*)(ws + off); off += NTOK;
  int* cnt   = (int*)(ws + off); off += NEXP;
  int* lists = (int*)(ws + off); off += (size_t)NEXP * NTOK;
  float* cq  = qkv;                        // reuse: qkv dead after self-attn
  float* ckv = qkv + (size_t)NTOK * DIMC;  // [B, 96(padded), 1024]

  dim3 b128(128);
  const float scale = 0.125f;  // hd^-0.5

  // ---- pre-convert weights to f16 ([N,K] layout everywhere) ----
  cvt16_k<<<dim3((3 * DIMC * DIMC) / 1024), dim3(256), 0, stream>>>(sa_in_w, saw16, (long)3 * DIMC * DIMC);
  cvt16_k<<<dim3((DIMC * DIMC) / 1024), dim3(256), 0, stream>>>(sa_out_w, saow16, (long)DIMC * DIMC);
  cvt16_k<<<dim3((3 * DIMC * DIMC) / 1024), dim3(256), 0, stream>>>(ca_in_w, caw16, (long)3 * DIMC * DIMC);
  cvt16_k<<<dim3((DIMC * DIMC) / 1024), dim3(256), 0, stream>>>(ca_out_w, caow16, (long)DIMC * DIMC);
  for (int e = 0; e < NEXP; ++e) {
    cvtT_k<<<dim3((DIMC * HIDD) / 256), dim3(256), 0, stream>>>(
        e_w1 + (size_t)e * DIMC * HIDD, ew1t + (size_t)e * HIDD * DIMC, DIMC, HIDD);
    cvtT_k<<<dim3((HIDD * DIMC) / 256), dim3(256), 0, stream>>>(
        e_w2 + (size_t)e * HIDD * DIMC, ew2t + (size_t)e * DIMC * HIDD, HIDD, DIMC);
  }

  // ---- styles + demodulated f16 1x1-conv weights ----
  style_k<<<dim3(BATCH), dim3(DIMC), 0, stream>>>(w, pin_mw, pin_mb, style_in);
  style_k<<<dim3(BATCH), dim3(DIMC), 0, stream>>>(w, pout_mw, pout_mb, style_out);
  modw_k<<<dim3(DIMC, BATCH), dim3(256), 0, stream>>>(pin_w, style_in, wt16_in);
  modw_k<<<dim3(DIMC, BATCH), dim3(256), 0, stream>>>(pout_w, style_out, wt16_out);

  // modconv-in: x_flat[b,t,o] = sum_i x[b,i,t] * wt_in[b,o,i]
  gemm_k<true, false, false, false, false, false><<<dim3(8, 16, BATCH), b128, 0, stream>>>(
      x, HW, (long)DIMC * HW, wt16_in, DIMC, (long)DIMC * DIMC, nullptr,
      x_flat, DIMC, (long)HW * DIMC, HW, DIMC, nullptr, nullptr);

  // self-attention
  ln_k<<<dim3(NTOK), b128, 0, stream>>>(x_flat, ln1_g, ln1_b, xn);
  gemm_k<false, false, false, false, false, false><<<dim3(24, 64, 1), b128, 0, stream>>>(
      xn, DIMC, 0, saw16, DIMC, 0, sa_in_b, qkv, 3 * DIMC, 0, NTOK, DIMC, nullptr, nullptr);
  attn_k<<<dim3(16, BATCH * HEADS), b128, 0, stream>>>(
      qkv, 3 * DIMC, (long)HW * 3 * DIMC, 0,
      qkv, 3 * DIMC, (long)HW * 3 * DIMC, DIMC,
      qkv, 3 * DIMC, (long)HW * 3 * DIMC, 2 * DIMC,
      attn_o, HW, scale);
  gemm_k<false, false, true, false, false, false><<<dim3(8, 64, 1), b128, 0, stream>>>(
      attn_o, DIMC, 0, saow16, DIMC, 0, sa_out_b, x_flat, DIMC, 0, NTOK, DIMC, nullptr, nullptr);

  // cross-attention (kv = 77 text tokens, masked)
  ln_k<<<dim3(NTOK), b128, 0, stream>>>(x_flat, ln2_g, ln2_b, xn);
  gemm_k<false, false, false, false, false, false><<<dim3(8, 64, 1), b128, 0, stream>>>(
      xn, DIMC, 0, caw16, DIMC, 0, ca_in_b, cq, DIMC, 0, NTOK, DIMC, nullptr, nullptr);
  gemm_k<false, false, false, false, false, false><<<dim3(16, 2, BATCH), b128, 0, stream>>>(
      text_seq, DIMC, (long)TXTLEN * DIMC, caw16 + (size_t)DIMC * DIMC, DIMC, 0, ca_in_b + DIMC,
      ckv, 2 * DIMC, (long)96 * 2 * DIMC, TXTLEN, DIMC, nullptr, nullptr);
  attn_k<<<dim3(16, BATCH * HEADS), b128, 0, stream>>>(
      cq, DIMC, (long)HW * DIMC, 0,
      ckv, 2 * DIMC, (long)96 * 2 * DIMC, 0,
      ckv, 2 * DIMC, (long)96 * 2 * DIMC, DIMC,
      attn_o, TXTLEN, scale);
  gemm_k<false, false, true, false, false, false><<<dim3(8, 64, 1), b128, 0, stream>>>(
      attn_o, DIMC, 0, caow16, DIMC, 0, ca_out_b, x_flat, DIMC, 0, NTOK, DIMC, nullptr, nullptr);

  // MoE: route, gather per-expert, run only the selected expert per token
  ln_k<<<dim3(NTOK), b128, 0, stream>>>(x_flat, ln3_g, ln3_b, xn);
  float* onehot = (float*)d_out + (size_t)BATCH * DIMC * HW;
  router_k<<<dim3(NTOK), b128, 0, stream>>>(xn, w, r_feat, r_text, r_comb, onehot, eid);
  zero8_k<<<dim3(1), dim3(32), 0, stream>>>(cnt);
  scatter_k<<<dim3(NTOK / 256), dim3(256), 0, stream>>>(eid, cnt, lists);
  for (int e = 0; e < NEXP; ++e) {
    // hid[0..cnt) = gelu(gather(xn) @ e_w1[e] + e_b1[e])
    gemm_k<false, true, false, false, true, false><<<dim3(HIDD / 64, NTOK / 64), b128, 0, stream>>>(
        xn, DIMC, 0, ew1t + (size_t)e * HIDD * DIMC, DIMC, 0, e_b1 + (size_t)e * HIDD,
        hid, HIDD, 0, NTOK, DIMC, cnt + e, lists + (size_t)e * NTOK);
    // x_flat[token] += hid @ e_w2[e] + e_b2[e]  (scatter)
    gemm_k<false, false, true, false, false, true><<<dim3(DIMC / 64, NTOK / 64), b128, 0, stream>>>(
        hid, HIDD, 0, ew2t + (size_t)e * DIMC * HIDD, HIDD, 0, e_b2 + (size_t)e * DIMC,
        x_flat, DIMC, 0, NTOK, HIDD, cnt + e, lists + (size_t)e * NTOK);
  }

  // modconv-out: out[b,o,t] = sum_i x_flat[b,t,i] * wt_out[b,o,i] (transposed store)
  gemm_k<false, false, false, true, false, false><<<dim3(8, 16, BATCH), b128, 0, stream>>>(
      x_flat, DIMC, (long)HW * DIMC, wt16_out, DIMC, (long)DIMC * HW, nullptr,
      (float*)d_out, HW, (long)DIMC * HW, HW, DIMC, nullptr, nullptr);

  (void)in_sizes; (void)n_in; (void)out_size; (void)ws_size;
}
